// fVSRN_39230231281694
// MI455X (gfx1250) — matrix operand order repaired and checked by reading, not compiled, not run on hardware
//
#include <hip/hip_runtime.h>
#include <hip/hip_bf16.h>

typedef __attribute__((ext_vector_type(16))) _Float16 v16h;
typedef __attribute__((ext_vector_type(8)))  _Float16 v8h;
typedef __attribute__((ext_vector_type(8)))  float    v8f;

#define GRES 96
#define NODES 128
#define IN_DIM 52              // C + L*ND*2 = 16 + 36
#define KIN_PAD 64             // pad 52 -> 64 for 2 K-tiles of 32
#define STRIDE 136             // LDS row pitch in halves: 272B, 16B aligned, conflict-free
#define WAVES_PER_BLOCK 4
#define PTS_PER_WAVE 16
#define PTS_PER_BLOCK (WAVES_PER_BLOCK * PTS_PER_WAVE)

// weight-fragment regions in d_ws (units: halves). frag(kt,nt,lane) = 16 contiguous halves.
#define OFF_WIN   0            // 2 ktiles * 8 ntiles * 32 lanes * 16 = 8192
#define OFF_WH0   8192         // 4*8*32*16 = 16384 each
#define OFF_WH1   24576
#define OFF_WH2   40960
#define WS_HALVES 57344        // * 2 bytes = 114,688 B of workspace

// ---------------------------------------------------------------------------
// Prep kernel: convert f32 weights -> f16 WMMA B-fragments (doc layout:
// lane l holds column n = l&15, K = 16*(i>>3) + 8*(l>>4) + (i&7) for i in 0..15).
// Zero-pads K beyond the real K (52 for w_in).
// ---------------------------------------------------------------------------
__global__ void fvsrn_prep_weights(const float* __restrict__ w_in,
                                   const float* __restrict__ w_h0,
                                   const float* __restrict__ w_h1,
                                   const float* __restrict__ w_h2,
                                   _Float16* __restrict__ ws) {
  int t = blockIdx.x * blockDim.x + threadIdx.x;
  if (t >= WS_HALVES) return;
  const float* src;
  int K, local;
  if (t < OFF_WH0)      { src = w_in; K = IN_DIM; local = t - OFF_WIN; }
  else if (t < OFF_WH1) { src = w_h0; K = NODES;  local = t - OFF_WH0; }
  else if (t < OFF_WH2) { src = w_h1; K = NODES;  local = t - OFF_WH1; }
  else                  { src = w_h2; K = NODES;  local = t - OFF_WH2; }
  int i    = local & 15;
  int lane = (local >> 4) & 31;
  int nt   = (local >> 9) & 7;
  int kt   = local >> 12;
  int k = kt * 32 + (i >> 3) * 16 + (lane >> 4) * 8 + (i & 7);   // B-fragment K striping
  int n = nt * 16 + (lane & 15);            // column
  float v = (k < K) ? src[k * NODES + n] : 0.0f;
  ws[t] = (_Float16)v;
}

// ---------------------------------------------------------------------------
// One MLP layer on a 16-row tile: D[16x128] = relu(A[16xK] * W[Kx128]).
// A fragments gathered from LDS (two 16B ds_load_b128 per fragment, ISA A
// layout: kbase = 8*(lane>>4), K in {kt*32+kbase+0..7, kt*32+16+kbase+0..7}).
// ---------------------------------------------------------------------------
template <int KTILES>
__device__ __forceinline__ void mlp_layer(const _Float16* __restrict__ src,
                                          _Float16* __restrict__ dst,
                                          const _Float16* __restrict__ wfrag,
                                          int lane) {
  const int col  = lane & 15;
  const int half = lane >> 4;
  const int kb   = half * 8;
  const _Float16* arow = src + (lane & 15) * STRIDE;
  for (int nt = 0; nt < 8; ++nt) {
    v8f acc = {};
#pragma unroll
    for (int kt = 0; kt < KTILES; ++kt) {
      v8h lo = *(const v8h*)(arow + kt * 32 + kb);
      v8h hi = *(const v8h*)(arow + kt * 32 + 16 + kb);
      v16h a = __builtin_shufflevector(lo, hi, 0, 1, 2, 3, 4, 5, 6, 7,
                                       8, 9, 10, 11, 12, 13, 14, 15);
      v16h b = *(const v16h*)(wfrag + (((kt * 8 + nt) * 32 + lane) << 4));
      acc = __builtin_amdgcn_wmma_f32_16x16x32_f16(
          /*neg_a=*/false, a, /*neg_b=*/false, b,
          /*c_mod=*/(short)0, acc, /*reuse_a=*/false, /*reuse_b=*/false);
    }
    // C layout: VGPR r -> (M = r + 8*half, N = col). ReLU + f16 + LDS store.
#pragma unroll
    for (int r = 0; r < 8; ++r) {
      float v = acc[r] > 0.0f ? acc[r] : 0.0f;
      dst[(r + half * 8) * STRIDE + nt * 16 + col] = (_Float16)v;
    }
  }
}

// ---------------------------------------------------------------------------
// Fused kernel: trilinear gather + positional encoding -> f16 tile in LDS,
// 4 WMMA layers (wave-private ping-pong LDS buffers), VALU output dot.
// ---------------------------------------------------------------------------
__global__ __launch_bounds__(PTS_PER_BLOCK * 2) void fvsrn_fused(
    const float* __restrict__ x,
    const float* __restrict__ fg,
    const float* __restrict__ w_out,
    const _Float16* __restrict__ ws,
    float* __restrict__ out,
    int n) {
  __shared__ _Float16 smem[WAVES_PER_BLOCK * 2 * 16 * STRIDE];

  const int tid  = threadIdx.x;
  const int wave = tid >> 5;
  const int lane = tid & 31;
  const int m    = lane & 15;       // row of the 16-point tile
  const int half = lane >> 4;       // half-wave: channel / K split

  _Float16* buf0 = smem + wave * (2 * 16 * STRIDE);
  _Float16* buf1 = buf0 + 16 * STRIDE;

  const int base = (blockIdx.x * WAVES_PER_BLOCK + wave) * PTS_PER_WAVE;
  int p = base + m;
  if (p >= n) p = n - 1;            // clamp; store is guarded later

  // ---- input features -> buf0[m][0..63] ----
  const float px = x[p * 3 + 0];
  const float py = x[p * 3 + 1];
  const float pz = x[p * 3 + 2];

  // trilinear (align_corners=True): x->W, y->H, z->D on [C, D, H, W]
  const float fx = (px + 1.0f) * 0.5f * (GRES - 1);
  const float fy = (py + 1.0f) * 0.5f * (GRES - 1);
  const float fz = (pz + 1.0f) * 0.5f * (GRES - 1);
  const float x0f = floorf(fx), y0f = floorf(fy), z0f = floorf(fz);
  const float tx = fx - x0f, ty = fy - y0f, tz = fz - z0f;
  int x0 = (int)x0f; x0 = x0 < 0 ? 0 : (x0 > GRES - 1 ? GRES - 1 : x0);
  int y0 = (int)y0f; y0 = y0 < 0 ? 0 : (y0 > GRES - 1 ? GRES - 1 : y0);
  int z0 = (int)z0f; z0 = z0 < 0 ? 0 : (z0 > GRES - 1 ? GRES - 1 : z0);
  const int x1 = x0 + 1 > GRES - 1 ? GRES - 1 : x0 + 1;
  const int y1 = y0 + 1 > GRES - 1 ? GRES - 1 : y0 + 1;
  const int z1 = z0 + 1 > GRES - 1 ? GRES - 1 : z0 + 1;
  const float w000 = (1 - tx) * (1 - ty) * (1 - tz);
  const float w001 = tx * (1 - ty) * (1 - tz);
  const float w010 = (1 - tx) * ty * (1 - tz);
  const float w011 = tx * ty * (1 - tz);
  const float w100 = (1 - tx) * (1 - ty) * tz;
  const float w101 = tx * (1 - ty) * tz;
  const float w110 = (1 - tx) * ty * tz;
  const float w111 = tx * ty * tz;
  const int i00 = (z0 * GRES + y0) * GRES;
  const int i01 = (z0 * GRES + y1) * GRES;
  const int i10 = (z1 * GRES + y0) * GRES;
  const int i11 = (z1 * GRES + y1) * GRES;

  const int cbase = half * 8;       // lanes 0-15: ch 0..7, lanes 16-31: ch 8..15
#pragma unroll
  for (int c = 0; c < 8; ++c) {
    const float* g = fg + (size_t)(cbase + c) * (GRES * GRES * GRES);
    float f = w000 * g[i00 + x0] + w001 * g[i00 + x1]
            + w010 * g[i01 + x0] + w011 * g[i01 + x1]
            + w100 * g[i10 + x0] + w101 * g[i10 + x1]
            + w110 * g[i11 + x0] + w111 * g[i11 + x1];
    buf0[m * STRIDE + 36 + cbase + c] = (_Float16)f;
  }

  if (lane < 16) {                  // positional encoding + K padding zeros
    const float crd[3] = {px, py, pz};
#pragma unroll
    for (int l = 0; l < 6; ++l) {
      const float freq = 3.14159265358979323846f * (float)(1 << l);
#pragma unroll
      for (int d = 0; d < 3; ++d) {
        const float a = crd[d] * freq;
        buf0[m * STRIDE + l * 6 + d]     = (_Float16)__sinf(a);
        buf0[m * STRIDE + l * 6 + 3 + d] = (_Float16)__cosf(a);
      }
    }
#pragma unroll
    for (int c = IN_DIM; c < KIN_PAD; ++c) buf0[m * STRIDE + c] = (_Float16)0.0f;
  }
  // wave-private LDS tile; LDS ops are in-order within a wave -> no barrier.

  // ---- 4 WMMA layers, ping-pong buffers ----
  mlp_layer<2>(buf0, buf1, ws + OFF_WIN, lane);   // 52(->64) -> 128
  mlp_layer<4>(buf1, buf0, ws + OFF_WH0, lane);   // 128 -> 128
  mlp_layer<4>(buf0, buf1, ws + OFF_WH1, lane);   // 128 -> 128
  mlp_layer<4>(buf1, buf0, ws + OFF_WH2, lane);   // 128 -> 128

  // ---- output layer: y = h4 . w_out (half-wave split over K, shfl-combine) --
  float partial = 0.0f;
  const _Float16* hrow = buf0 + m * STRIDE + half * 64;
  const float* wo = w_out + half * 64;
#pragma unroll
  for (int c = 0; c < 64; c += 8) {
    v8h hv = *(const v8h*)(hrow + c);
#pragma unroll
    for (int j = 0; j < 8; ++j)
      partial = fmaf((float)hv[j], wo[c + j], partial);
  }
  const float other = __shfl_xor(partial, 16, 32);
  const float y = partial + other;
  const int idx = base + m;
  if (lane < 16 && idx < n)
    out[idx] = y * (1.0f - 0.0f) + 0.0f;          // VOL_MAX/VOL_MIN scale
}

// ---------------------------------------------------------------------------
extern "C" void kernel_launch(void* const* d_in, const int* in_sizes, int n_in,
                              void* d_out, int out_size, void* d_ws, size_t ws_size,
                              hipStream_t stream) {
  const float* x    = (const float*)d_in[0];
  const float* fg   = (const float*)d_in[1];
  const float* w_in = (const float*)d_in[2];
  const float* w_h0 = (const float*)d_in[3];
  const float* w_h1 = (const float*)d_in[4];
  const float* w_h2 = (const float*)d_in[5];
  const float* wout = (const float*)d_in[6];
  float* out = (float*)d_out;
  _Float16* ws = (_Float16*)d_ws;   // needs 114,688 B

  const int n = in_sizes[0] / 3;

  fvsrn_prep_weights<<<(WS_HALVES + 255) / 256, 256, 0, stream>>>(
      w_in, w_h0, w_h1, w_h2, ws);

  const int blocks = (n + PTS_PER_BLOCK - 1) / PTS_PER_BLOCK;
  fvsrn_fused<<<blocks, PTS_PER_BLOCK * 2, 0, stream>>>(
      x, fg, wout, ws, out, n);
}
